// GraphAttentionLayer_38036230373822
// MI455X (gfx1250) — matrix operand order repaired and checked by reading, not compiled, not run on hardware
//
#include <hip/hip_runtime.h>

// GAT layer for MI455X (gfx1250, wave32).
// Memory floor: adj = 268MB @ 23.3TB/s ~= 11.5us -> stream adj once with an
// online-softmax (flash-attention style); P@H on v_wmma_f32_16x16x32_f16.
// h kept in f32 for scores, f16 (transposed) as WMMA B operand (L2-resident).

#define GN    8192
#define FIN   256
#define FOUT  64
#define LRELU 0.2f
#define MASKV (-9000000000000000.0f)
#define SPLITS 8            // waves per 16-row group (j-range split)
#define JT    (GN / SPLITS) // 1024 columns per wave
#define JTILE 32            // K per WMMA step

typedef __attribute__((ext_vector_type(16))) _Float16 v16h;
typedef __attribute__((ext_vector_type(4)))  _Float16 v4h;
typedef __attribute__((ext_vector_type(8)))  float    v8f;
typedef __attribute__((ext_vector_type(4)))  int      v4i;

// ---------------------------------------------------------------------------
// Kernel 1: h = input@W + b_lin (f32), s1 = h@a1, s2 = h@a2 (f32),
//           hT[f][i] = (f16)h[i][f]  (transposed for WMMA B-operand loads)
// One block of 256 threads per 16 rows.
// ---------------------------------------------------------------------------
__global__ __launch_bounds__(256) void gat_h_kernel(
    const float* __restrict__ input, const float* __restrict__ W,
    const float* __restrict__ b_lin, const float* __restrict__ a,
    _Float16* __restrict__ hT, float* __restrict__ s1, float* __restrict__ s2)
{
    __shared__ float ls_in[16 * FIN];   // 16 rows x 256 (16KB)
    __shared__ float ls_h[16][FOUT + 1]; // padded to kill bank conflicts

    const int t  = threadIdx.x;
    const int i0 = blockIdx.x * 16;

    // Stage the 16x256 input tile (coalesced float4).
    {
        const float4* src = (const float4*)(input + (size_t)i0 * FIN);
        float4* dst = (float4*)ls_in;
#pragma unroll
        for (int seg = 0; seg < 4; ++seg)
            dst[seg * 256 + t] = src[seg * 256 + t];
    }
    __syncthreads();

    const int f  = t & 63;   // output feature
    const int rg = t >> 6;   // row group of 4 rows
    float acc0 = b_lin[f], acc1 = acc0, acc2 = acc0, acc3 = acc0;
    const float* inrow = ls_in + rg * 4 * FIN;
    for (int kk = 0; kk < FIN; ++kk) {
        float w = W[kk * FOUT + f];              // coalesced, reused 4x
        acc0 = fmaf(inrow[0 * FIN + kk], w, acc0); // LDS broadcast reads
        acc1 = fmaf(inrow[1 * FIN + kk], w, acc1);
        acc2 = fmaf(inrow[2 * FIN + kk], w, acc2);
        acc3 = fmaf(inrow[3 * FIN + kk], w, acc3);
    }
    ls_h[rg * 4 + 0][f] = acc0;
    ls_h[rg * 4 + 1][f] = acc1;
    ls_h[rg * 4 + 2][f] = acc2;
    ls_h[rg * 4 + 3][f] = acc3;
    __syncthreads();

    // Transposed f16 store: thread -> (feature fo, 4 consecutive node ids).
    {
        const int fo = t >> 2, seg = t & 3;
        v4h v;
#pragma unroll
        for (int j = 0; j < 4; ++j)
            v[j] = (_Float16)ls_h[seg * 4 + j][fo];
        *(v4h*)(hT + (size_t)fo * GN + i0 + seg * 4) = v; // 8B aligned
    }

    // s1/s2 from full-precision h.
    if (t < 16) {
        float v1 = 0.f, v2 = 0.f;
        for (int ff = 0; ff < FOUT; ++ff) {
            float hv = ls_h[t][ff];
            v1 = fmaf(hv, a[ff], v1);
            v2 = fmaf(hv, a[FOUT + ff], v2);
        }
        s1[i0 + t] = v1;
        s2[i0 + t] = v2;
    }
}

// ---------------------------------------------------------------------------
// Kernel 2: online-softmax attention + P@H with WMMA f16 (f32 accumulate).
// Block = 256 threads = 8 waves; block owns 16 rows, wave w streams columns
// [w*1024, (w+1)*1024) in 32-wide tiles, states merged through LDS.
// A-domain: lane covers row (lane&15); element e -> K = e + half*16.
// C-domain: lane,reg r -> row r + half*8, col (lane&15)+16*ntile.
// Accumulator rescale is vote-guarded: after the running max saturates,
// alpha == 1.0 exactly and the whole rescale (8 bpermutes + 32 muls) is
// skipped via a wave-uniform branch.
// ---------------------------------------------------------------------------
__global__ __launch_bounds__(256) void gat_attn_kernel(
    const int* __restrict__ adj, const _Float16* __restrict__ hT,
    const float* __restrict__ s1, const float* __restrict__ s2,
    const float* __restrict__ bias, float* __restrict__ out)
{
    __shared__ float ls_acc[SPLITS][16][FOUT]; // 32KB
    __shared__ float ls_m[SPLITS][16];
    __shared__ float ls_l[SPLITS][16];
    __shared__ float ls_scale[SPLITS][16];
    __shared__ float ls_rden[16];

    const int lane = threadIdx.x & 31;
    const int wv   = threadIdx.x >> 5;
    const int half = lane >> 4;
    const int rA   = lane & 15;          // row this lane feeds into A
    const int row0 = blockIdx.x * 16;
    const int jb   = wv * JT;

    const float s1row = s1[row0 + rA];
    float m = -INFINITY, lsum = 0.f;
    v8f c0 = {}, c1 = {}, c2 = {}, c3 = {};

    const int* adjrow = adj + (size_t)(row0 + rA) * GN;
    const _Float16* hTl = hT + (size_t)rA * GN;

    for (int j0 = jb; j0 < jb + JT; j0 += JTILE) {
        // gfx1250 speculative (SCOPE_SYS) prefetch of next adj tile; OOB at
        // the tail is silently dropped, so no guard branch needed.
        __builtin_prefetch(adjrow + j0 + JTILE + half * 16, 0, 0);

        const int b0i = j0 + half * 16;       // e = 0..7  -> K = half*16 + e
        const int b1i = j0 + half * 16 + 8;   // e = 8..15 -> K = half*16 + e
        v4i ad0 = __builtin_nontemporal_load((const v4i*)(adjrow + b0i));
        v4i ad1 = __builtin_nontemporal_load((const v4i*)(adjrow + b0i + 4));
        v4i ad2 = __builtin_nontemporal_load((const v4i*)(adjrow + b1i));
        v4i ad3 = __builtin_nontemporal_load((const v4i*)(adjrow + b1i + 4));
        float4 t0 = *(const float4*)(s2 + b0i);
        float4 t1 = *(const float4*)(s2 + b0i + 4);
        float4 t2 = *(const float4*)(s2 + b1i);
        float4 t3 = *(const float4*)(s2 + b1i + 4);

        // B operands: 32 contiguous bytes of hT per lane (K = e + half*16).
        v16h bm0 = *(const v16h*)(hTl + (size_t)0 * 16 * GN + j0 + half * 16);
        v16h bm1 = *(const v16h*)(hTl + (size_t)1 * 16 * GN + j0 + half * 16);
        v16h bm2 = *(const v16h*)(hTl + (size_t)2 * 16 * GN + j0 + half * 16);
        v16h bm3 = *(const v16h*)(hTl + (size_t)3 * 16 * GN + j0 + half * 16);

        float ev[16];
        int   am[16];
        ev[0]=t0.x; ev[1]=t0.y; ev[2]=t0.z; ev[3]=t0.w;
        ev[4]=t1.x; ev[5]=t1.y; ev[6]=t1.z; ev[7]=t1.w;
        ev[8]=t2.x; ev[9]=t2.y; ev[10]=t2.z; ev[11]=t2.w;
        ev[12]=t3.x; ev[13]=t3.y; ev[14]=t3.z; ev[15]=t3.w;
        am[0]=ad0[0]; am[1]=ad0[1]; am[2]=ad0[2]; am[3]=ad0[3];
        am[4]=ad1[0]; am[5]=ad1[1]; am[6]=ad1[2]; am[7]=ad1[3];
        am[8]=ad2[0]; am[9]=ad2[1]; am[10]=ad2[2]; am[11]=ad2[3];
        am[12]=ad3[0]; am[13]=ad3[1]; am[14]=ad3[2]; am[15]=ad3[3];

        float tmax = -INFINITY;
#pragma unroll
        for (int e = 0; e < 16; ++e) {
            float v = s1row + ev[e];
            v = (v > 0.f) ? v : (LRELU * v);     // leaky_relu
            v = (am[e] > 0) ? v : MASKV;         // adjacency mask
            ev[e] = v;
            tmax = fmaxf(tmax, v);
        }
        const float rowmax = fmaxf(tmax, __shfl_xor(tmax, 16)); // both K-halves

        // Wave-uniform vote: rescale only if some row's max actually grew.
        if (__ballot(rowmax > m)) {
            const float newm = fmaxf(m, rowmax);
            const float alpha = __expf(m - newm);
            m = newm;
            lsum *= alpha;
#pragma unroll
            for (int r = 0; r < 8; ++r) {        // broadcast alpha to C layout
                float ar = __shfl(alpha, r + (half << 3));
                c0[r] *= ar; c1[r] *= ar; c2[r] *= ar; c3[r] *= ar;
            }
        }

        float psum = 0.f;
        v16h av;
#pragma unroll
        for (int e = 0; e < 16; ++e) {
            float p = __expf(ev[e] - m);
            psum += p;
            av[e] = (_Float16)p;
        }
        psum += __shfl_xor(psum, 16);
        lsum += psum;

        c0 = __builtin_amdgcn_wmma_f32_16x16x32_f16(false, av, false, bm0,
                                                    (short)0, c0, false, false);
        c1 = __builtin_amdgcn_wmma_f32_16x16x32_f16(false, av, false, bm1,
                                                    (short)0, c1, false, false);
        c2 = __builtin_amdgcn_wmma_f32_16x16x32_f16(false, av, false, bm2,
                                                    (short)0, c2, false, false);
        c3 = __builtin_amdgcn_wmma_f32_16x16x32_f16(false, av, false, bm3,
                                                    (short)0, c3, false, false);
    }

    // Dump per-wave state.
    if (lane < 16) { ls_m[wv][lane] = m; ls_l[wv][lane] = lsum; }
#pragma unroll
    for (int r = 0; r < 8; ++r) {
        const int row = r + half * 8;
        ls_acc[wv][row][rA + 0]  = c0[r];
        ls_acc[wv][row][rA + 16] = c1[r];
        ls_acc[wv][row][rA + 32] = c2[r];
        ls_acc[wv][row][rA + 48] = c3[r];
    }
    __syncthreads();

    // Per-row merge factors across the SPLITS partial softmaxes.
    if (threadIdx.x < 16) {
        const int row = threadIdx.x;
        float M = -INFINITY;
        for (int w = 0; w < SPLITS; ++w) M = fmaxf(M, ls_m[w][row]);
        float den = 0.f;
        for (int w = 0; w < SPLITS; ++w) {
            float sc = __expf(ls_m[w][row] - M);
            ls_scale[w][row] = sc;
            den = fmaf(sc, ls_l[w][row], den);
        }
        ls_rden[row] = 1.0f / den;
    }
    __syncthreads();

    // Final combine: 16x64 outputs, 4 per thread, coalesced stores.
#pragma unroll
    for (int k = 0; k < 4; ++k) {
        const int elem = threadIdx.x + k * 256;
        const int row = elem >> 6, col = elem & 63;
        float v = 0.f;
        for (int w = 0; w < SPLITS; ++w)
            v = fmaf(ls_scale[w][row], ls_acc[w][row][col], v);
        out[(size_t)(row0 + row) * FOUT + col] = v * ls_rden[row] + bias[col];
    }
}

extern "C" void kernel_launch(void* const* d_in, const int* in_sizes, int n_in,
                              void* d_out, int out_size, void* d_ws, size_t ws_size,
                              hipStream_t stream) {
    const float* input = (const float*)d_in[0];
    const int*   adj   = (const int*)d_in[1];
    const float* W     = (const float*)d_in[2];
    const float* b_lin = (const float*)d_in[3];
    const float* a     = (const float*)d_in[4];
    const float* bias  = (const float*)d_in[5];
    float* out = (float*)d_out;

    char* ws = (char*)d_ws;
    _Float16* hT = (_Float16*)ws;                           // 64*8192 f16 = 1MB
    float* s1 = (float*)(ws + (size_t)FOUT * GN * sizeof(_Float16));
    float* s2 = s1 + GN;

    gat_h_kernel<<<GN / 16, 256, 0, stream>>>(input, W, b_lin, a, hT, s1, s2);
    gat_attn_kernel<<<GN / 16, 256, 0, stream>>>(adj, hT, s1, s2, bias, out);
}
